// DrugGNN_74277164417510
// MI455X (gfx1250) — matrix operand order repaired and checked by reading, not compiled, not run on hardware
//
#include <hip/hip_runtime.h>
#include <hip/hip_bf16.h>
#include <cstdint>
#include <cstddef>

// ---------------- problem constants (match reference) ----------------
#define NN 200000
#define EE 800000
#define GG 4096
// DIM_DRUG = 78 (padded to 96 for K%32==0), GNN = 128

typedef __bf16 bf16_t;
typedef bf16_t v16bf  __attribute__((ext_vector_type(16)));
typedef float  v8f    __attribute__((ext_vector_type(8)));
typedef unsigned uint4v __attribute__((ext_vector_type(4)));

union V16 { v16bf v; uint4v q[2]; };

// ---------------- small utility kernels ----------------
__global__ void k_zero_f32(float* p, int n) {
    int i = blockIdx.x * blockDim.x + threadIdx.x;
    if (i < n) p[i] = 0.0f;
}

// eval-mode BatchNorm over the raw 78-dim drug features
__global__ void k_bn0(const float* __restrict__ x,
                      const float* __restrict__ g, const float* __restrict__ b,
                      const float* __restrict__ m, const float* __restrict__ v,
                      float* __restrict__ y) {
    int i = blockIdx.x * blockDim.x + threadIdx.x;
    if (i >= NN * 78) return;
    int c = i % 78;
    y[i] = (x[i] - m[c]) * rsqrtf(v[c] + 1e-5f) * g[c] + b[c];
}

// edge scatter-add: agg[dst] += x[src]  (78-dim f32)
__global__ void k_scatter78(const float* __restrict__ x, const int* __restrict__ ei,
                            float* __restrict__ agg) {
    long long i = (long long)blockIdx.x * blockDim.x + threadIdx.x;
    if (i >= (long long)EE * 78) return;
    int e = (int)(i / 78), c = (int)(i % 78);
    int s = ei[e], d = ei[EE + e];
    atomicAdd(&agg[(long long)d * 78 + c], x[(long long)s * 78 + c]);
}

// edge scatter-add, 128-dim, gather from bf16 (halved read bytes), accumulate f32
__global__ void k_scatter128(const bf16_t* __restrict__ h, const int* __restrict__ ei,
                             float* __restrict__ agg) {
    long long i = (long long)blockIdx.x * blockDim.x + threadIdx.x;
    if (i >= (long long)EE * 128) return;
    int e = (int)(i >> 7), c = (int)(i & 127);
    int s = ei[e], d = ei[EE + e];
    atomicAdd(&agg[((long long)d << 7) + c], (float)h[((long long)s << 7) + c]);
}

// A1[i,k] = bf16( xbn[i,k] + agg1[i,k] ), zero-padded to K=96
__global__ void k_buildA1(const float* __restrict__ x, const float* __restrict__ agg,
                          bf16_t* __restrict__ A) {
    int i = blockIdx.x * blockDim.x + threadIdx.x;
    if (i >= NN * 96) return;
    int n = i / 96, k = i % 96;
    float val = 0.0f;
    if (k < 78) val = x[n * 78 + k] + agg[n * 78 + k];
    A[i] = (bf16_t)val;
}

// o = bf16(a + b)  (used for h1 + agg2 -> A2)
__global__ void k_add_to_bf16(const float* __restrict__ a, const float* __restrict__ b,
                              bf16_t* __restrict__ o, int n) {
    int i = blockIdx.x * blockDim.x + threadIdx.x;
    if (i < n) o[i] = (bf16_t)(a[i] + b[i]);
}

// in-degree over dst edges (self-loop handled in the transform below)
__global__ void k_deg(const int* __restrict__ ei, float* __restrict__ deg) {
    int e = blockIdx.x * blockDim.x + threadIdx.x;
    if (e < EE) atomicAdd(&deg[ei[EE + e]], 1.0f);
}
__global__ void k_dinv(float* d) {
    int i = blockIdx.x * blockDim.x + threadIdx.x;
    if (i < NN) d[i] = rsqrtf(d[i] + 1.0f);   // +1 for self loop; deg>=1 so no zero check
}

// transpose + convert all 4 GIN weights to bf16 [OUT, K] (K padded to 96 for layer-1a)
__global__ void k_prep_w(const float* __restrict__ w1, const float* __restrict__ w2,
                         const float* __restrict__ w3, const float* __restrict__ w4,
                         bf16_t* W1T, bf16_t* W2T, bf16_t* W3T, bf16_t* W4T) {
    int i = blockIdx.x * blockDim.x + threadIdx.x;
    if (i < 256 * 96) { int o = i / 96, k = i % 96;
        W1T[i] = (k < 78) ? (bf16_t)w1[k * 256 + o] : (bf16_t)0.0f; return; }
    i -= 256 * 96;
    if (i < 128 * 256) { int o = i / 256, k = i % 256; W2T[i] = (bf16_t)w2[k * 128 + o]; return; }
    i -= 128 * 256;
    if (i < 256 * 128) { int o = i / 128, k = i % 128; W3T[i] = (bf16_t)w3[k * 256 + o]; return; }
    i -= 256 * 128;
    if (i < 128 * 256) { int o = i / 256, k = i % 256; W4T[i] = (bf16_t)w4[k * 128 + o]; return; }
}

// ---------------- WMMA GEMM: C[N,OUT] = epilogue(A[N,K]@W[K,OUT] + bias) ----------------
// One wave computes a 16x64 strip: 4 accumulators, K-loop step 32.
// A fragment per ISA 16-bit A layout: lane l -> row (l&15),
//   K halves {0..7,16..23} (lanes 0-15) / {8..15,24..31} (lanes 16-31): two b128 loads.
// B from pre-transposed Wt[OUT,K]: lane l -> col (l&15), same K halves as A: two b128 loads.
__global__ __launch_bounds__(256) void k_gemm_bf16(
    const bf16_t* __restrict__ A, int ldA, int K,
    const bf16_t* __restrict__ Wt, const float* __restrict__ bias, int OUT,
    float* __restrict__ outF, bf16_t* __restrict__ outB,
    const float* __restrict__ bng, const float* __restrict__ bnb,
    const float* __restrict__ bnm, const float* __restrict__ bnv)
{
    const int colGroups = OUT >> 6;                       // 64 output cols per wave
    int wave = (blockIdx.x * blockDim.x + threadIdx.x) >> 5;
    int lane = threadIdx.x & 31;
    int rowTile = wave / colGroups;
    if (rowTile >= NN / 16) return;                       // wave-uniform
    int colBase = (wave % colGroups) << 6;
    int rowBase = rowTile << 4;

    int m     = lane & 15;
    int aBase = (lane >> 4) << 3;                         // 0 or 8
    int bBase = (lane >> 4) << 3;                         // 0 or 8
    int nIdx  = lane & 15;

    const bf16_t* aRow = A + (size_t)(rowBase + m) * ldA;
    v8f acc[4] = {};

    for (int kk = 0; kk < K; kk += 32) {
        V16 a;
        a.q[0] = *(const uint4v*)(aRow + kk + aBase);
        a.q[1] = *(const uint4v*)(aRow + kk + aBase + 16);
        const bf16_t* bp0 = Wt + (size_t)(colBase + nIdx) * K + kk + bBase;
#pragma unroll
        for (int j = 0; j < 4; ++j) {
            V16 b;
            const bf16_t* bp = bp0 + (size_t)(16 * j) * K;
            b.q[0] = *(const uint4v*)(bp);
            b.q[1] = *(const uint4v*)(bp + 16);
            acc[j] = __builtin_amdgcn_wmma_f32_16x16x32_bf16(
                false, a.v, false, b.v, (short)0, acc[j], false, false);
        }
    }

    // epilogue: +bias, ReLU, optional eval-BN, store f32 and/or bf16
    int mTop = (lane >> 4) << 3;
#pragma unroll
    for (int j = 0; j < 4; ++j) {
        int col = colBase + j * 16 + nIdx;
        float bs = bias[col];
        float sg = 1.0f, sb = 0.0f;
        if (bng) { sg = bng[col] * rsqrtf(bnv[col] + 1e-5f); sb = bnb[col] - bnm[col] * sg; }
#pragma unroll
        for (int r = 0; r < 8; ++r) {
            int row = rowBase + mTop + r;
            float v = acc[j][r] + bs;
            v = fmaxf(v, 0.0f);
            if (bng) v = v * sg + sb;
            size_t oi = (size_t)row * OUT + col;
            if (outF) outF[oi] = v;
            if (outB) outB[oi] = (bf16_t)v;
        }
    }
}

// ---------------- SAG-pool scoring / segment softmax / pooling ----------------
__device__ inline unsigned fkey(float f) {
    unsigned u = __float_as_uint(f);
    return (u & 0x80000000u) ? ~u : (u | 0x80000000u);   // order-preserving key
}
__device__ inline float funkey(unsigned k) {
    unsigned u = (k & 0x80000000u) ? (k & 0x7FFFFFFFu) : ~k;
    return __uint_as_float(u);
}

// one wave per node: s = h[i].dot(w);  P = gcn_b + s*dinv^2 (self loop term)
__global__ void k_score_init(const float* __restrict__ h, const float* __restrict__ w,
                             const float* __restrict__ bptr, const float* __restrict__ dinv,
                             float* __restrict__ S, float* __restrict__ P) {
    int wave = (blockIdx.x * blockDim.x + threadIdx.x) >> 5;
    int lane = threadIdx.x & 31;
    if (wave >= NN) return;
    const float* hr = h + (size_t)wave * 128;
    float s = hr[lane] * w[lane] + hr[lane + 32] * w[lane + 32]
            + hr[lane + 64] * w[lane + 64] + hr[lane + 96] * w[lane + 96];
    for (int off = 16; off; off >>= 1) s += __shfl_xor(s, off, 32);
    if (lane == 0) {
        S[wave] = s;
        float di = dinv[wave];
        P[wave] = bptr[0] + s * di * di;
    }
}

__global__ void k_edge_prop(const int* __restrict__ ei, const float* __restrict__ S,
                            const float* __restrict__ dinv, float* __restrict__ P) {
    int e = blockIdx.x * blockDim.x + threadIdx.x;
    if (e >= EE) return;
    int s = ei[e], d = ei[EE + e];
    atomicAdd(&P[d], S[s] * dinv[s] * dinv[d]);
}

__global__ void k_tanh_max(const float* __restrict__ P, float* __restrict__ S,
                           const int* __restrict__ batch, unsigned* __restrict__ gmax) {
    int i = blockIdx.x * blockDim.x + threadIdx.x;
    if (i >= NN) return;
    float t = tanhf(P[i]);
    S[i] = t;
    atomicMax(&gmax[batch[i]], fkey(t));
}

__global__ void k_exp_sum(const float* __restrict__ S, const int* __restrict__ batch,
                          const unsigned* __restrict__ gmax, float* __restrict__ P,
                          float* __restrict__ gsum) {
    int i = blockIdx.x * blockDim.x + threadIdx.x;
    if (i >= NN) return;
    float e = __expf(S[i] - funkey(gmax[batch[i]]));
    P[i] = e;
    atomicAdd(&gsum[batch[i]], e);
}

__global__ void k_pool(const float* __restrict__ h, const float* __restrict__ P,
                       const float* __restrict__ gsum, const int* __restrict__ batch,
                       float* __restrict__ out, int off) {
    long long idx = (long long)blockIdx.x * blockDim.x + threadIdx.x;
    if (idx >= (long long)NN * 128) return;
    int i = (int)(idx >> 7), c = (int)(idx & 127);
    int g = batch[i];
    float w = P[i] / gsum[g];
    atomicAdd(&out[(size_t)g * 256 + off + c], h[idx] * w);
}

// ---------------- host-side orchestration ----------------
static inline int GRID(long long n) { return (int)((n + 255) / 256); }

extern "C" void kernel_launch(void* const* d_in, const int* in_sizes, int n_in,
                              void* d_out, int out_size, void* d_ws, size_t ws_size,
                              hipStream_t stream) {
    (void)in_sizes; (void)n_in; (void)out_size;
    const float* drug = (const float*)d_in[0];
    const int*   ei   = (const int*)  d_in[1];
    const int*   ib   = (const int*)  d_in[2];
    const float* bn0g = (const float*)d_in[3],  *bn0b = (const float*)d_in[4];
    const float* bn0m = (const float*)d_in[5],  *bn0v = (const float*)d_in[6];
    const float* bn1g = (const float*)d_in[7],  *bn1b = (const float*)d_in[8];
    const float* bn1m = (const float*)d_in[9],  *bn1v = (const float*)d_in[10];
    const float* bn2g = (const float*)d_in[11], *bn2b = (const float*)d_in[12];
    const float* bn2m = (const float*)d_in[13], *bn2v = (const float*)d_in[14];
    const float* g1w1 = (const float*)d_in[15], *g1b1 = (const float*)d_in[16];
    const float* g1w2 = (const float*)d_in[17], *g1b2 = (const float*)d_in[18];
    const float* g2w1 = (const float*)d_in[19], *g2b1 = (const float*)d_in[20];
    const float* g2w2 = (const float*)d_in[21], *g2b2 = (const float*)d_in[22];
    const float* gc1w = (const float*)d_in[23], *gc1b = (const float*)d_in[24];
    const float* gc2w = (const float*)d_in[25], *gc2b = (const float*)d_in[26];
    float* out = (float*)d_out;
    char*  ws  = (char*)d_ws;

    // scratch layout (all offsets 256B aligned); AGG2 aliases the dead XBN+AGG1 region
    size_t o = 0;
    const size_t XBN  = o; o += (size_t)NN * 78 * 4;       // 62.4 MB
    const size_t AGG1 = o; o += (size_t)NN * 78 * 4;       // 62.4 MB
    const size_t AGG2 = XBN;                               // needs N*128*4 <= 124.8 MB ✓
    const size_t A1   = o; o += (size_t)NN * 96 * 2;       // 38.4 MB bf16
    const size_t HA   = o; o += (size_t)NN * 256 * 2;      // 102.4 MB bf16 (both layers)
    const size_t H1F  = o; o += (size_t)NN * 128 * 4;      // 102.4 MB
    const size_t H1B  = o; o += (size_t)NN * 128 * 2;      // 51.2 MB bf16
    const size_t A2   = o; o += (size_t)NN * 128 * 2;      // 51.2 MB bf16
    const size_t H2F  = o; o += (size_t)NN * 128 * 4;      // 102.4 MB
    const size_t DINV = o; o += (size_t)NN * 4;
    const size_t SBUF = o; o += (size_t)NN * 4;
    const size_t PBUF = o; o += (size_t)NN * 4;
    const size_t GMAX = o; o += (size_t)GG * 4;
    const size_t GSUM = o; o += (size_t)GG * 4;
    const size_t W1T  = o; o += (size_t)256 * 96 * 2;
    const size_t W2T  = o; o += (size_t)128 * 256 * 2;
    const size_t W3T  = o; o += (size_t)256 * 128 * 2;
    const size_t W4T  = o; o += (size_t)128 * 256 * 2;
    if (ws_size < o) return;  // scratch too small; deterministic no-op

    float*  xbn  = (float*)(ws + XBN);
    float*  agg1 = (float*)(ws + AGG1);
    float*  agg2 = (float*)(ws + AGG2);
    bf16_t* a1   = (bf16_t*)(ws + A1);
    bf16_t* ha   = (bf16_t*)(ws + HA);
    float*  h1f  = (float*)(ws + H1F);
    bf16_t* h1b  = (bf16_t*)(ws + H1B);
    bf16_t* a2   = (bf16_t*)(ws + A2);
    float*  h2f  = (float*)(ws + H2F);
    float*  dinv = (float*)(ws + DINV);
    float*  sbuf = (float*)(ws + SBUF);
    float*  pbuf = (float*)(ws + PBUF);
    unsigned* gmax = (unsigned*)(ws + GMAX);
    float*  gsum = (float*)(ws + GSUM);
    bf16_t* w1t = (bf16_t*)(ws + W1T);
    bf16_t* w2t = (bf16_t*)(ws + W2T);
    bf16_t* w3t = (bf16_t*)(ws + W3T);
    bf16_t* w4t = (bf16_t*)(ws + W4T);

    // ---- prep / zeroing ----
    k_prep_w<<<GRID(256*96 + 3*128*256), 256, 0, stream>>>(g1w1, g1w2, g2w1, g2w2, w1t, w2t, w3t, w4t);
    k_zero_f32<<<GRID((long long)GG*2*128), 256, 0, stream>>>(out, GG*2*128);
    k_zero_f32<<<GRID((long long)NN*78), 256, 0, stream>>>(agg1, NN*78);
    k_zero_f32<<<GRID(NN), 256, 0, stream>>>(dinv, NN);
    k_zero_f32<<<GRID(2*GG), 256, 0, stream>>>((float*)gmax, 2*GG);  // gmax key 0 < key(-1); gsum=0

    // ---- layer 1: BN0 -> GIN1 (WMMA) -> ReLU+BN1 ----
    k_bn0<<<GRID((long long)NN*78), 256, 0, stream>>>(drug, bn0g, bn0b, bn0m, bn0v, xbn);
    k_scatter78<<<GRID((long long)EE*78), 256, 0, stream>>>(xbn, ei, agg1);
    k_buildA1<<<GRID((long long)NN*96), 256, 0, stream>>>(xbn, agg1, a1);
    // [N,96]@[96,256] -> ReLU -> HA (bf16)
    k_gemm_bf16<<<(NN/16)*(256/64)/8 * 1, 256, 0, stream>>>(a1, 96, 96, w1t, g1b1, 256,
        nullptr, ha, nullptr, nullptr, nullptr, nullptr);
    // [N,256]@[256,128] +b -> ReLU -> BN1 -> h1 (f32 + bf16)
    k_gemm_bf16<<<(NN/16)*(128/64)/8, 256, 0, stream>>>(ha, 256, 256, w2t, g1b2, 128,
        h1f, h1b, bn1g, bn1b, bn1m, bn1v);

    // ---- degree normalization (shared by both SAG pools) ----
    k_deg<<<GRID(EE), 256, 0, stream>>>(ei, dinv);
    k_dinv<<<GRID(NN), 256, 0, stream>>>(dinv);

    // ---- SAG pool 1 on h1 ----
    k_score_init<<<GRID((long long)NN*32), 256, 0, stream>>>(h1f, gc1w, gc1b, dinv, sbuf, pbuf);
    k_edge_prop<<<GRID(EE), 256, 0, stream>>>(ei, sbuf, dinv, pbuf);
    k_tanh_max<<<GRID(NN), 256, 0, stream>>>(pbuf, sbuf, ib, gmax);
    k_exp_sum<<<GRID(NN), 256, 0, stream>>>(sbuf, ib, gmax, pbuf, gsum);
    k_pool<<<GRID((long long)NN*128), 256, 0, stream>>>(h1f, pbuf, gsum, ib, out, 0);

    // ---- layer 2: GIN2 (WMMA) -> ReLU+BN2 ----
    k_zero_f32<<<GRID((long long)NN*128), 256, 0, stream>>>(agg2, NN*128);
    k_scatter128<<<GRID((long long)EE*128), 256, 0, stream>>>(h1b, ei, agg2);
    k_add_to_bf16<<<GRID((long long)NN*128), 256, 0, stream>>>(h1f, agg2, a2, NN*128);
    k_gemm_bf16<<<(NN/16)*(256/64)/8, 256, 0, stream>>>(a2, 128, 128, w3t, g2b1, 256,
        nullptr, ha, nullptr, nullptr, nullptr, nullptr);
    k_gemm_bf16<<<(NN/16)*(128/64)/8, 256, 0, stream>>>(ha, 256, 256, w4t, g2b2, 128,
        h2f, nullptr, bn2g, bn2b, bn2m, bn2v);

    // ---- SAG pool 2 on h2 ----
    k_zero_f32<<<GRID(2*GG), 256, 0, stream>>>((float*)gmax, 2*GG);
    k_score_init<<<GRID((long long)NN*32), 256, 0, stream>>>(h2f, gc2w, gc2b, dinv, sbuf, pbuf);
    k_edge_prop<<<GRID(EE), 256, 0, stream>>>(ei, sbuf, dinv, pbuf);
    k_tanh_max<<<GRID(NN), 256, 0, stream>>>(pbuf, sbuf, ib, gmax);
    k_exp_sum<<<GRID(NN), 256, 0, stream>>>(sbuf, ib, gmax, pbuf, gsum);
    k_pool<<<GRID((long long)NN*128), 256, 0, stream>>>(h2f, pbuf, gsum, ib, out, 128);
}
